// GatedAttention_44452911514129
// MI455X (gfx1250) — matrix operand order repaired and checked by reading, not compiled, not run on hardware
//
#include <hip/hip_runtime.h>

typedef __attribute__((ext_vector_type(16))) _Float16 v16h;
typedef __attribute__((ext_vector_type(8)))  _Float16 h8;
typedef __attribute__((ext_vector_type(8)))  float    v8f;
typedef __attribute__((ext_vector_type(4)))  unsigned int v4u;
typedef __attribute__((ext_vector_type(4)))  int      v4i;
typedef __attribute__((ext_vector_type(8)))  int      v8i;

#define B_   2
#define T_   2048
#define C_   2048
#define H_   16
#define HKV_ 4
#define D_   128

union FragH { v16h v; h8 c[2]; _Float16 h[16]; };

// ---------------------------------------------------------------------------
// TDM: async-load a (tile_rows x 32 halves) tile from a row-major f16 matrix
// (row stride = K halves) into LDS with a 40-half padded row stride
// (pad_interval = 16 DWORDs, pad_amount = 4 DWORDs). Tracked by TENSORcnt.
// ---------------------------------------------------------------------------
__device__ __forceinline__ void tdm_load_tile(unsigned lds_off,
                                              const _Float16* gsrc,
                                              unsigned tile_rows, unsigned K) {
    unsigned long long ga = (unsigned long long)(void*)gsrc;
    v4u g0;
    g0[0] = 1u;                                   // count=1, user descriptor
    g0[1] = lds_off;                              // lds_addr (bytes)
    g0[2] = (unsigned)ga;                         // global_addr[31:0]
    g0[3] = (unsigned)((ga >> 32) & 0x01FFFFFFu)  // global_addr[56:32]
          | (2u << 30);                           // type=2 ("image")
    const unsigned td1 = 1u << 20;                // tensor_dim1 (ample)
    v8i g1;
    g1[0] = (int)((1u << 16)                      // data_size = 2 bytes
                | (1u << 20)                      // pad_enable
                | (3u << 22)                      // pad_interval: 16 DWORDs
                | (3u << 25));                    // pad_amount:   4 DWORDs
    g1[1] = (int)((K & 0xFFFFu) << 16);           // tensor_dim0[15:0]
    g1[2] = (int)(((K >> 16) & 0xFFFFu)           // tensor_dim0[31:16]
                | ((td1 & 0xFFFFu) << 16));       // tensor_dim1[15:0]
    g1[3] = (int)(((td1 >> 16) & 0xFFFFu)         // tensor_dim1[31:16]
                | (32u << 16));                   // tile_dim0 = 32 halves
    g1[4] = (int)(tile_rows & 0xFFFFu);           // tile_dim1; tile_dim2=0
    g1[5] = (int)K;                               // tensor_dim0_stride[31:0]
    g1[6] = 0;                                    // stride hi / dim1_stride
    g1[7] = 0;
    v4i z4 = {};
#if __clang_major__ >= 23
    v8i z8 = {};
    __builtin_amdgcn_tensor_load_to_lds(g0, g1, z4, z4, z8, 0);
#else
    __builtin_amdgcn_tensor_load_to_lds(g0, g1, z4, z4, 0);
#endif
}

// ---------------------------------------------------------------------------
// C(MxN,f32) = Ah(MxK,f16) * BT(NxK,f16)^T. 256 threads = 8 waves (2M x 4N),
// wave tile 32x64, block tile 64x256, K-step 32, TDM double-buffered LDS.
// ---------------------------------------------------------------------------
__global__ __launch_bounds__(256)
void gemm_tdm_kernel(const _Float16* __restrict__ Ah,
                     const _Float16* __restrict__ BT,
                     float* __restrict__ Cm, int M, int N, int K) {
    __shared__ _Float16 As[2][64 * 40];
    __shared__ _Float16 Bs[2][256 * 40];
    (void)M;

    const int tid  = threadIdx.x;
    const int lane = tid & 31;
    const int wid  = tid >> 5;
    const int wm   = wid >> 2;          // 0..1
    const int wn   = wid & 3;           // 0..3
    const int col  = lane & 15;
    const int hi   = lane >> 4;
    const int m_base = blockIdx.y * 64;
    const int n_base = blockIdx.x * 256;

    v8f z = {};
    v8f acc[2][4];
#pragma unroll
    for (int s2 = 0; s2 < 2; ++s2)
#pragma unroll
        for (int f = 0; f < 4; ++f) acc[s2][f] = z;

    const _Float16* Ab = Ah + (size_t)m_base * K;
    const _Float16* Bb = BT + (size_t)n_base * K;

    if (wid == 0) {
        tdm_load_tile((unsigned)(unsigned long long)(void*)&As[0][0], Ab, 64, K);
        tdm_load_tile((unsigned)(unsigned long long)(void*)&Bs[0][0], Bb, 256, K);
    }

    const int nsteps = K >> 5;
    for (int s = 0; s < nsteps; ++s) {
        const int cur = s & 1;
        if (wid == 0) {
            if (s + 1 < nsteps) {
                tdm_load_tile((unsigned)(unsigned long long)(void*)&As[cur ^ 1][0],
                              Ab + (size_t)(s + 1) * 32, 64, K);
                tdm_load_tile((unsigned)(unsigned long long)(void*)&Bs[cur ^ 1][0],
                              Bb + (size_t)(s + 1) * 32, 256, K);
                __builtin_amdgcn_s_wait_tensorcnt(2);   // current buffer done
            } else {
                __builtin_amdgcn_s_wait_tensorcnt(0);
            }
        }
        __syncthreads();

        FragH a[2];
#pragma unroll
        for (int s2 = 0; s2 < 2; ++s2) {
            int row = wm * 32 + s2 * 16 + col;
            a[s2].c[0] = *(const h8*)&As[cur][row * 40 + hi * 8];
            a[s2].c[1] = *(const h8*)&As[cur][row * 40 + 16 + hi * 8];
        }
#pragma unroll
        for (int f = 0; f < 4; ++f) {
            FragH bf;
            int n = wn * 64 + f * 16 + col;
            bf.c[0] = *(const h8*)&Bs[cur][n * 40 + hi * 8];
            bf.c[1] = *(const h8*)&Bs[cur][n * 40 + 16 + hi * 8];
            acc[0][f] = __builtin_amdgcn_wmma_f32_16x16x32_f16(
                false, a[0].v, false, bf.v, (short)0, acc[0][f], false, false);
            acc[1][f] = __builtin_amdgcn_wmma_f32_16x16x32_f16(
                false, a[1].v, false, bf.v, (short)0, acc[1][f], false, false);
        }
        __syncthreads();
    }

#pragma unroll
    for (int s2 = 0; s2 < 2; ++s2)
#pragma unroll
        for (int f = 0; f < 4; ++f)
#pragma unroll
            for (int r = 0; r < 8; ++r) {
                int mrow = m_base + wm * 32 + s2 * 16 + r + hi * 8;
                int ncol = n_base + wn * 64 + f * 16 + col;
                Cm[(size_t)mrow * N + ncol] = acc[s2][f][r];
            }
}

// ---------------------------------------------------------------------------
// Elementwise f32 -> f16 convert.
// ---------------------------------------------------------------------------
__global__ void cvt_f32_f16_kernel(const float* __restrict__ s,
                                   _Float16* __restrict__ d, int n) {
    int i = blockIdx.x * blockDim.x + threadIdx.x;
    if (i < n) d[i] = (_Float16)s[i];
}

// ---------------------------------------------------------------------------
// W (KxN, f32) -> WT (NxK, f16). 32x32 tiles via LDS.
// ---------------------------------------------------------------------------
__global__ __launch_bounds__(256)
void transpose_cvt_kernel(const float* __restrict__ W, _Float16* __restrict__ WT,
                          int K, int N) {
    __shared__ float tile[32][33];
    const int n0 = blockIdx.x * 32;
    const int k0 = blockIdx.y * 32;
    const int tx = threadIdx.x & 31;
    const int ty = threadIdx.x >> 5;        // 0..7
#pragma unroll
    for (int i = 0; i < 32; i += 8)
        tile[ty + i][tx] = W[(size_t)(k0 + ty + i) * N + n0 + tx];
    __syncthreads();
#pragma unroll
    for (int i = 0; i < 32; i += 8)
        WT[(size_t)(n0 + ty + i) * K + k0 + tx] = (_Float16)tile[tx][ty + i];
}

// ---------------------------------------------------------------------------
// RoPE (full head-dim, half-split rotate) + f32->f16 convert.
// ---------------------------------------------------------------------------
__global__ __launch_bounds__(64)
void rope_cvt_kernel(const float* __restrict__ src, _Float16* __restrict__ dst,
                     int nheads) {
    int gid = blockIdx.x;
    int row = gid / nheads;
    int hh  = gid - row * nheads;
    int t   = row & (T_ - 1);
    int d   = threadIdx.x;          // 0..63
    size_t base = (size_t)row * nheads * D_ + (size_t)hh * D_;
    float u1 = src[base + d];
    float u2 = src[base + d + 64];
    float inv = __expf(-(float)d * (9.210340371976184f / 64.0f)); // 10000^(-d/64)
    float th  = (float)t * inv;
    float s, c;
    __sincosf(th, &s, &c);
    dst[base + d]      = (_Float16)(u1 * c - u2 * s);
    dst[base + d + 64] = (_Float16)(u2 * c + u1 * s);
}

// ---------------------------------------------------------------------------
// Flash-attention (causal, GQA) + sigmoid gate. One wave per 16 query rows.
// ---------------------------------------------------------------------------
__global__ __launch_bounds__(32)
void attn_kernel(const _Float16* __restrict__ Qh, const _Float16* __restrict__ Kh,
                 const _Float16* __restrict__ Vh, const float* __restrict__ Gf,
                 float* __restrict__ AO) {
    __shared__ _Float16 Pl[16 * 32];

    const int lane = threadIdx.x & 31;
    const int col  = lane & 15;
    const int hi   = lane >> 4;
    const int q0   = blockIdx.x * 16;
    const int h    = blockIdx.y;
    const int b    = blockIdx.z;
    const int hkv  = h >> 2;
    const float scale = 0.08838834764831845f;   // 1/sqrt(128)

    FragH qa[4];
    {
        const _Float16* qp = Qh + ((size_t)(b * T_ + q0 + col)) * (H_ * D_) + h * D_;
#pragma unroll
        for (int i = 0; i < 4; ++i) {
            qa[i].c[0] = *(const h8*)(qp + i * 32 + hi * 8);
            qa[i].c[1] = *(const h8*)(qp + i * 32 + 16 + hi * 8);
        }
    }

    v8f z = {};
    v8f O[8];
#pragma unroll
    for (int j = 0; j < 8; ++j) O[j] = z;
    float mr[8], lr[8];
#pragma unroll
    for (int r = 0; r < 8; ++r) { mr[r] = -1e30f; lr[r] = 0.f; }

    const int ntiles = (q0 + 47) >> 5;
    for (int ti = 0; ti < ntiles; ++ti) {
        const int k0 = ti * 32;

        v8f S0 = z, S1 = z;
        {
            const _Float16* kp0 = Kh + ((size_t)(b * T_ + k0 + col)) * (HKV_ * D_) + hkv * D_;
            const _Float16* kp1 = kp0 + (size_t)16 * (HKV_ * D_);
#pragma unroll
            for (int i = 0; i < 4; ++i) {
                FragH kb;
                kb.c[0] = *(const h8*)(kp0 + i * 32 + hi * 8);
                kb.c[1] = *(const h8*)(kp0 + i * 32 + 16 + hi * 8);
                S0 = __builtin_amdgcn_wmma_f32_16x16x32_f16(
                    false, qa[i].v, false, kb.v, (short)0, S0, false, false);
                kb.c[0] = *(const h8*)(kp1 + i * 32 + hi * 8);
                kb.c[1] = *(const h8*)(kp1 + i * 32 + 16 + hi * 8);
                S1 = __builtin_amdgcn_wmma_f32_16x16x32_f16(
                    false, qa[i].v, false, kb.v, (short)0, S1, false, false);
            }
        }

#pragma unroll
        for (int r = 0; r < 8; ++r) {
            int rowq = q0 + r + hi * 8;
            float s0 = S0[r] * scale;
            float s1 = S1[r] * scale;
            if (k0 + col > rowq)      s0 = -3.0e38f;
            if (k0 + 16 + col > rowq) s1 = -3.0e38f;
            float tm = fmaxf(s0, s1);
#pragma unroll
            for (int mk = 1; mk < 16; mk <<= 1) tm = fmaxf(tm, __shfl_xor(tm, mk, 16));
            float mn = fmaxf(mr[r], tm);
            float p0 = __expf(s0 - mn);
            float p1 = __expf(s1 - mn);
            float alpha = __expf(mr[r] - mn);
            float ps = p0 + p1;
#pragma unroll
            for (int mk = 1; mk < 16; mk <<= 1) ps += __shfl_xor(ps, mk, 16);
            lr[r] = lr[r] * alpha + ps;
            mr[r] = mn;
#pragma unroll
            for (int j = 0; j < 8; ++j) O[j][r] = O[j][r] * alpha;
            Pl[(r + hi * 8) * 32 + col]      = (_Float16)p0;
            Pl[(r + hi * 8) * 32 + 16 + col] = (_Float16)p1;
        }
        __syncthreads();

        FragH pa;
        pa.c[0] = *(const h8*)&Pl[col * 32 + hi * 16];
        pa.c[1] = *(const h8*)&Pl[col * 32 + hi * 16 + 8];
        const _Float16* vp = Vh + ((size_t)(b * T_ + k0 + hi * 16)) * (HKV_ * D_)
                                + hkv * D_ + col;
#pragma unroll
        for (int j = 0; j < 8; ++j) {
            FragH vb;
#pragma unroll
            for (int e = 0; e < 16; ++e)
                vb.h[e] = vp[(size_t)e * (HKV_ * D_) + j * 16];
            O[j] = __builtin_amdgcn_wmma_f32_16x16x32_f16(
                false, pa.v, false, vb.v, (short)0, O[j], false, false);
        }
        __syncthreads();
    }

#pragma unroll
    for (int j = 0; j < 8; ++j)
#pragma unroll
        for (int r = 0; r < 8; ++r) {
            int row = q0 + r + hi * 8;
            size_t idx = ((size_t)(b * T_ + row)) * (H_ * D_) + h * D_ + j * 16 + col;
            float g = Gf[idx];
            float val = (O[j][r] / lr[r]) * (1.0f / (1.0f + __expf(-g)));
            AO[idx] = val;
        }
}

// ---------------------------------------------------------------------------
extern "C" void kernel_launch(void* const* d_in, const int* in_sizes, int n_in,
                              void* d_out, int out_size, void* d_ws, size_t ws_size,
                              hipStream_t stream) {
    (void)in_sizes; (void)n_in; (void)out_size; (void)ws_size;
    const float* x  = (const float*)d_in[0];
    const float* Wq = (const float*)d_in[1];
    const float* Wk = (const float*)d_in[2];
    const float* Wv = (const float*)d_in[3];
    const float* Wg = (const float*)d_in[4];
    const float* Wo = (const float*)d_in[5];
    float* out = (float*)d_out;

    char* w = (char*)d_ws;
    _Float16* xh  = (_Float16*)(w);                    // 16 MB [M, C] f16
    _Float16* WqT = (_Float16*)(w + 16777216ull);      //  8 MB [2048, 2048]
    _Float16* WkT = (_Float16*)(w + 25165824ull);      //  2 MB [512, 2048]
    _Float16* WvT = (_Float16*)(w + 27262976ull);      //  2 MB
    _Float16* WgT = (_Float16*)(w + 29360128ull);      //  8 MB
    _Float16* WoT = (_Float16*)(w + 37748736ull);      //  8 MB
    float*    Qf  = (float*)(w + 46137344ull);         // 32 MB [M, H*D] f32
    float*    Kf  = (float*)(w + 79691776ull);         //  8 MB
    float*    Vf  = (float*)(w + 88080384ull);         //  8 MB
    float*    Gf  = (float*)(w + 96468992ull);         // 32 MB
    _Float16* Qh  = (_Float16*)(w + 130023424ull);     // 16 MB f16 (post-RoPE)
    _Float16* Kh  = (_Float16*)(w + 146800640ull);     //  4 MB
    _Float16* Vh  = (_Float16*)(w + 150994944ull);     //  4 MB
    float*    AO  = Qf;                                // alias: Qf dead after RoPE
    _Float16* AOh = xh;                                // alias: xh dead after projections

    const int M = B_ * T_;                             // 4096
    dim3 blk(256);

    // one-time precision/layout conversion (bandwidth-trivial)
    int nx = M * C_;
    cvt_f32_f16_kernel<<<(nx + 255) / 256, 256, 0, stream>>>(x, xh, nx);
    transpose_cvt_kernel<<<dim3((H_*D_)/32,  C_/32), 256, 0, stream>>>(Wq, WqT, C_, H_*D_);
    transpose_cvt_kernel<<<dim3((HKV_*D_)/32, C_/32), 256, 0, stream>>>(Wk, WkT, C_, HKV_*D_);
    transpose_cvt_kernel<<<dim3((HKV_*D_)/32, C_/32), 256, 0, stream>>>(Wv, WvT, C_, HKV_*D_);
    transpose_cvt_kernel<<<dim3((H_*D_)/32,  C_/32), 256, 0, stream>>>(Wg, WgT, C_, H_*D_);
    transpose_cvt_kernel<<<dim3(C_/32, (H_*D_)/32), 256, 0, stream>>>(Wo, WoT, H_*D_, C_);

    // projections (TDM-fed WMMA GEMMs)
    gemm_tdm_kernel<<<dim3((H_*D_)/256,  M/64), blk, 0, stream>>>(xh, WqT, Qf, M, H_*D_,  C_);
    gemm_tdm_kernel<<<dim3((HKV_*D_)/256, M/64), blk, 0, stream>>>(xh, WkT, Kf, M, HKV_*D_, C_);
    gemm_tdm_kernel<<<dim3((HKV_*D_)/256, M/64), blk, 0, stream>>>(xh, WvT, Vf, M, HKV_*D_, C_);
    gemm_tdm_kernel<<<dim3((H_*D_)/256,  M/64), blk, 0, stream>>>(xh, WgT, Gf, M, H_*D_,  C_);

    // RoPE(Q,K) + f16 convert; convert V
    rope_cvt_kernel<<<M * H_,   64, 0, stream>>>(Qf, Qh, H_);
    rope_cvt_kernel<<<M * HKV_, 64, 0, stream>>>(Kf, Kh, HKV_);
    int nv = M * HKV_ * D_;
    cvt_f32_f16_kernel<<<(nv + 255) / 256, 256, 0, stream>>>(Vf, Vh, nv);

    // causal GQA flash attention + sigmoid gate
    attn_kernel<<<dim3(T_ / 16, H_, B_), dim3(32), 0, stream>>>(Qh, Kh, Vh, Gf, AO);

    // output projection
    int nao = M * H_ * D_;
    cvt_f32_f16_kernel<<<(nao + 255) / 256, 256, 0, stream>>>(AO, AOh, nao);
    gemm_tdm_kernel<<<dim3(C_/256, M/64), blk, 0, stream>>>(AOh, WoT, out, M, C_, H_*D_);
}
